// MeasureEmbeddingNetwork_2293512536312
// MI455X (gfx1250) — matrix operand order repaired and checked by reading, not compiled, not run on hardware
//
#include <hip/hip_runtime.h>
#include <hip/hip_bf16.h>
#include <math.h>

typedef __attribute__((ext_vector_type(16))) _Float16 v16h;
typedef __attribute__((ext_vector_type(8)))  _Float16 v8h;
typedef __attribute__((ext_vector_type(8)))  float    v8f;

#define RIS_RCS 0.008862269254527581f
#define VOX_RCS 0.8862269254527581f
#define BN_EPS  1e-5f

#define WMMA(a,b,c) __builtin_amdgcn_wmma_f32_16x16x32_f16(false,(a),false,(b),(short)0,(c),false,false)

// ---------------- fragment loaders (CDNA5 16-bit WMMA layouts, ISA 7.12.2) ----------------
// A (16x32, MxK): lane m=lane&15, kg=lane>>4; elements 0..7 -> K = kg*8+e, 8..15 -> K = 16+kg*8+(e-8)
__device__ __forceinline__ v16h frag_a(const _Float16* base, int ld, int row0, int k0, int lane) {
  int m = lane & 15, kg = lane >> 4;
  const _Float16* p = base + (size_t)(row0 + m) * ld + k0 + kg * 8;
  v8h lo = *(const v8h*)p;
  v8h hi = *(const v8h*)(p + 16);
  v16h r;
#pragma unroll
  for (int i = 0; i < 8; ++i) { r[i] = lo[i]; r[i + 8] = hi[i]; }
  return r;
}
// B (32x16, KxN) from a column-major "BT[col][k]" buffer: lane n=lane&15, kg=lane>>4; elements 0..7 -> K = k0+kg*8+e, 8..15 -> K = k0+16+kg*8+(e-8)
__device__ __forceinline__ v16h frag_b(const _Float16* base, int ld, int col0, int k0, int lane) {
  int n = lane & 15, kg = lane >> 4;
  const _Float16* p = base + (size_t)(col0 + n) * ld + k0 + kg * 8;
  v8h lo = *(const v8h*)p;
  v8h hi = *(const v8h*)(p + 16);
  v16h r;
#pragma unroll
  for (int i = 0; i < 8; ++i) { r[i] = lo[i]; r[i + 8] = hi[i]; }
  return r;
}

// ---------------- small pack / init kernels ----------------
__global__ void k_init(const float2* uc5, float* feat, float* stats) {
  int b = blockIdx.x, j = threadIdx.x;           // 256 x 256
  float2 c = uc5[j];                             // j = t*16+r
  feat[b * 512 + j]       = c.x;
  feat[b * 512 + 256 + j] = c.y;
  if (b == 0 && j < 16) stats[j] = 0.f;
}

__global__ void k_ris(const float* phase, float* rr, float* ri, int n) {
  int i = blockIdx.x * blockDim.x + threadIdx.x;
  if (i < n) { float p = phase[i]; rr[i] = cosf(p); ri[i] = -sinf(p); }  // e^{-i phi}
}

__global__ void k_xs(const float* x, float* xs, int n) {
  int i = blockIdx.x * blockDim.x + threadIdx.x;
  if (i < n) xs[i] = x[i] * VOX_RCS;
}

__global__ void k_pack_u4(const float2* uc4, _Float16* vr, _Float16* vi, _Float16* nr, _Float16* ni) {
  int i = blockIdx.x * 256 + threadIdx.x;        // 4096*1024
  int v = i >> 10, n = i & 1023;
  float2 c = uc4[i];
  vr[i] = (_Float16)c.x; vi[i] = (_Float16)c.y;
  nr[(size_t)n * 4096 + v] = (_Float16)c.x;
  ni[(size_t)n * 4096 + v] = (_Float16)c.y;
}

__global__ void k_pack_bt(const float2* src, _Float16* dr, _Float16* di, int rows) {
  int i = blockIdx.x * 256 + threadIdx.x;        // rows*16, src[row][r]
  if (i >= rows * 16) return;
  int row = i >> 4, r = i & 15;
  float2 c = src[i];
  dr[(size_t)r * rows + row] = (_Float16)c.x;
  di[(size_t)r * rows + row] = (_Float16)c.y;
}

__global__ void k_f32h(const float* s, _Float16* d, int n) {
  int i = blockIdx.x * blockDim.x + threadIdx.x;
  if (i < n) d[i] = (_Float16)s[i];
}

// ---------------- stage 1: S1 = A x uc4^T, fused (*Xv) x uc3 epilogue + Bm x uc3 ----------------
__global__ __launch_bounds__(256) void k_s1(
    const float2* uc0, const float2* uc1, const float* ris_r, const float* ris_i,
    const float* xs, const _Float16* u4v_r, const _Float16* u4v_i,
    const _Float16* u3t_r, const _Float16* u3t_i, float* feat) {
  __shared__ _Float16 shAr[16 * 512], shAi[16 * 512];   // 32 KB
  __shared__ _Float16 stgR[8 * 512], stgI[8 * 512];     // 16 KB
  const int b = blockIdx.y, chunk = blockIdx.x;
  const int tid = threadIdx.x, wave = tid >> 5, lane = tid & 31;
  const int vbase = chunk * 256 + wave * 32;            // wave owns v-pair [vbase, vbase+32)
  v8f crp0 = {}, crm0 = {}, ci0 = {}, crp1 = {}, crm1 = {}, ci1 = {};
  for (int pass = 0; pass < 2; ++pass) {                // K = 1024 in two 512 chunks
    __syncthreads();
    for (int i = tid; i < 8192; i += 256) {             // A = uc0 * ris * RIS_RCS (f16)
      int t = i >> 9, nl = i & 511, n = pass * 512 + nl;
      float2 c0 = uc0[t * 1024 + n];
      float rr = ris_r[b * 1024 + n], ri = ris_i[b * 1024 + n];
      shAr[t * 512 + nl] = (_Float16)((c0.x * rr - c0.y * ri) * RIS_RCS);
      shAi[t * 512 + nl] = (_Float16)((c0.x * ri + c0.y * rr) * RIS_RCS);
    }
    __syncthreads();
    for (int kk = 0; kk < 512; kk += 32) {
      int kg_ = pass * 512 + kk;
      v16h ar = frag_a(shAr, 512, 0, kk, lane);
      v16h ai = frag_a(shAi, 512, 0, kk, lane);
      v16h br0 = frag_b(u4v_r, 1024, vbase,      kg_, lane);
      v16h bi0 = frag_b(u4v_i, 1024, vbase,      kg_, lane);
      v16h br1 = frag_b(u4v_r, 1024, vbase + 16, kg_, lane);
      v16h bi1 = frag_b(u4v_i, 1024, vbase + 16, kg_, lane);
      crp0 = WMMA(ar, br0, crp0); crm0 = WMMA(ai, bi0, crm0);
      ci0  = WMMA(ar, bi0, ci0);  ci0  = WMMA(ai, br0, ci0);
      crp1 = WMMA(ar, br1, crp1); crm1 = WMMA(ai, bi1, crm1);
      ci1  = WMMA(ar, bi1, ci1);  ci1  = WMMA(ai, br1, ci1);
    }
  }
  const int kg = lane >> 4, cl = lane & 15;
  {  // scale columns by Xv (real) and restage as f16 A-layout (16 x 32, K = local v)
    float sx0 = xs[b * 4096 + vbase + cl];
    float sx1 = xs[b * 4096 + vbase + 16 + cl];
#pragma unroll
    for (int j = 0; j < 8; ++j) {
      int t = j + kg * 8;
      stgR[wave * 512 + t * 32 + cl]      = (_Float16)((crp0[j] - crm0[j]) * sx0);
      stgI[wave * 512 + t * 32 + cl]      = (_Float16)(ci0[j] * sx0);
      stgR[wave * 512 + t * 32 + 16 + cl] = (_Float16)((crp1[j] - crm1[j]) * sx1);
      stgI[wave * 512 + t * 32 + 16 + cl] = (_Float16)(ci1[j] * sx1);
    }
  }
  v16h sar = frag_a(stgR + wave * 512, 32, 0, 0, lane);
  v16h sai = frag_a(stgI + wave * 512, 32, 0, 0, lane);
  v16h u3r = frag_b(u3t_r, 4096, 0, vbase, lane);
  v16h u3i = frag_b(u3t_i, 4096, 0, vbase, lane);
  v8f rp = {}, rm = {}, rc = {};
  rp = WMMA(sar, u3r, rp); rm = WMMA(sai, u3i, rm);
  rc = WMMA(sar, u3i, rc); rc = WMMA(sai, u3r, rc);
  {  // fold in h_tx_roi_rx piece for this v-pair: Bm x uc3
    v16h bmr, bmi;
    int m = lane & 15;
#pragma unroll
    for (int e = 0; e < 16; ++e) {
      int koff = (e & 7) + ((e >> 3) * 16) + kg * 8;
      int v = vbase + koff;
      float2 c1 = uc1[m * 4096 + v];
      float s = xs[b * 4096 + v];
      bmr[e] = (_Float16)(c1.x * s);
      bmi[e] = (_Float16)(c1.y * s);
    }
    rp = WMMA(bmr, u3r, rp); rm = WMMA(bmi, u3i, rm);
    rc = WMMA(bmr, u3i, rc); rc = WMMA(bmi, u3r, rc);
  }
#pragma unroll
  for (int j = 0; j < 8; ++j) {
    int t = j + kg * 8;
    atomicAdd(&feat[b * 512 + t * 16 + cl],       rp[j] - rm[j]);
    atomicAdd(&feat[b * 512 + 256 + t * 16 + cl], rc[j]);
  }
}

// ---------------- stage 2: S2 = Bm x uc4, fused (*ris*RCS) x uc2 epilogue + A x uc2 ----------------
__global__ __launch_bounds__(256) void k_s2(
    const float2* uc0, const float2* uc1, const float* ris_r, const float* ris_i,
    const float* xs, const _Float16* u4n_r, const _Float16* u4n_i,
    const _Float16* u2t_r, const _Float16* u2t_i, float* feat) {
  __shared__ _Float16 shBr[16 * 512], shBi[16 * 512];
  __shared__ _Float16 stgR[8 * 512], stgI[8 * 512];
  const int b = blockIdx.y, chunk = blockIdx.x;
  const int tid = threadIdx.x, wave = tid >> 5, lane = tid & 31;
  const int nbase = chunk * 256 + wave * 32;
  v8f crp0 = {}, crm0 = {}, ci0 = {}, crp1 = {}, crm1 = {}, ci1 = {};
  for (int pass = 0; pass < 8; ++pass) {                // K = 4096 in eight 512 chunks
    __syncthreads();
    for (int i = tid; i < 8192; i += 256) {             // Bm = uc1 * (x*VOX_RCS)
      int t = i >> 9, vl = i & 511, v = pass * 512 + vl;
      float2 c1 = uc1[t * 4096 + v];
      float s = xs[b * 4096 + v];
      shBr[t * 512 + vl] = (_Float16)(c1.x * s);
      shBi[t * 512 + vl] = (_Float16)(c1.y * s);
    }
    __syncthreads();
    for (int kk = 0; kk < 512; kk += 32) {
      int kg_ = pass * 512 + kk;
      v16h ar = frag_a(shBr, 512, 0, kk, lane);
      v16h ai = frag_a(shBi, 512, 0, kk, lane);
      v16h br0 = frag_b(u4n_r, 4096, nbase,      kg_, lane);
      v16h bi0 = frag_b(u4n_i, 4096, nbase,      kg_, lane);
      v16h br1 = frag_b(u4n_r, 4096, nbase + 16, kg_, lane);
      v16h bi1 = frag_b(u4n_i, 4096, nbase + 16, kg_, lane);
      crp0 = WMMA(ar, br0, crp0); crm0 = WMMA(ai, bi0, crm0);
      ci0  = WMMA(ar, bi0, ci0);  ci0  = WMMA(ai, br0, ci0);
      crp1 = WMMA(ar, br1, crp1); crm1 = WMMA(ai, bi1, crm1);
      ci1  = WMMA(ar, bi1, ci1);  ci1  = WMMA(ai, br1, ci1);
    }
  }
  const int kg = lane >> 4, cl = lane & 15;
  {  // complex scale columns by ris * RIS_RCS, restage as f16 A-layout
    int n0 = nbase + cl, n1 = nbase + 16 + cl;
    float rr0 = ris_r[b * 1024 + n0], ri0 = ris_i[b * 1024 + n0];
    float rr1 = ris_r[b * 1024 + n1], ri1 = ris_i[b * 1024 + n1];
#pragma unroll
    for (int j = 0; j < 8; ++j) {
      int t = j + kg * 8;
      float s2r0 = crp0[j] - crm0[j], s2i0 = ci0[j];
      stgR[wave * 512 + t * 32 + cl]      = (_Float16)((s2r0 * rr0 - s2i0 * ri0) * RIS_RCS);
      stgI[wave * 512 + t * 32 + cl]      = (_Float16)((s2r0 * ri0 + s2i0 * rr0) * RIS_RCS);
      float s2r1 = crp1[j] - crm1[j], s2i1 = ci1[j];
      stgR[wave * 512 + t * 32 + 16 + cl] = (_Float16)((s2r1 * rr1 - s2i1 * ri1) * RIS_RCS);
      stgI[wave * 512 + t * 32 + 16 + cl] = (_Float16)((s2r1 * ri1 + s2i1 * rr1) * RIS_RCS);
    }
  }
  v16h sar = frag_a(stgR + wave * 512, 32, 0, 0, lane);
  v16h sai = frag_a(stgI + wave * 512, 32, 0, 0, lane);
  v16h u2r = frag_b(u2t_r, 1024, 0, nbase, lane);
  v16h u2i = frag_b(u2t_i, 1024, 0, nbase, lane);
  v8f rp = {}, rm = {}, rc = {};
  rp = WMMA(sar, u2r, rp); rm = WMMA(sai, u2i, rm);
  rc = WMMA(sar, u2i, rc); rc = WMMA(sai, u2r, rc);
  {  // fold in h_tx_ris_rx piece for this n-pair: A x uc2
    v16h afr, afi;
    int m = lane & 15;
#pragma unroll
    for (int e = 0; e < 16; ++e) {
      int koff = (e & 7) + ((e >> 3) * 16) + kg * 8;
      int n = nbase + koff;
      float2 c0 = uc0[m * 1024 + n];
      float rr = ris_r[b * 1024 + n], ri = ris_i[b * 1024 + n];
      afr[e] = (_Float16)((c0.x * rr - c0.y * ri) * RIS_RCS);
      afi[e] = (_Float16)((c0.x * ri + c0.y * rr) * RIS_RCS);
    }
    rp = WMMA(afr, u2r, rp); rm = WMMA(afi, u2i, rm);
    rc = WMMA(afr, u2i, rc); rc = WMMA(afi, u2r, rc);
  }
#pragma unroll
  for (int j = 0; j < 8; ++j) {
    int t = j + kg * 8;
    atomicAdd(&feat[b * 512 + t * 16 + cl],       rp[j] - rm[j]);
    atomicAdd(&feat[b * 512 + 256 + t * 16 + cl], rc[j]);
  }
}

// ---------------- generic f16 WMMA GEMM: C[M,N] = A[M,K] @ W[N,K]^T + bias ----------------
__global__ __launch_bounds__(256) void k_gemm(
    const _Float16* A, const _Float16* Bt, const float* bias, float* C, int N, int K) {
  int wave = threadIdx.x >> 5, lane = threadIdx.x & 31;
  int row0 = blockIdx.x * 16;
  int col0 = blockIdx.y * 128 + wave * 16;
  v8f acc = {};
  for (int k0 = 0; k0 < K; k0 += 32) {
    v16h a  = frag_a(A, K, row0, k0, lane);
    v16h bf = frag_b(Bt, K, col0, k0, lane);
    acc = WMMA(a, bf, acc);
  }
  int kg = lane >> 4, n = col0 + (lane & 15);
  float bv = bias[n];
#pragma unroll
  for (int j = 0; j < 8; ++j)
    C[(size_t)(row0 + j + kg * 8) * N + n] = acc[j] + bv;
}

// ---------------- global batch-norm (scalar stats) ----------------
__global__ void k_bnstats(const float* x, int n, float* st) {
  float s = 0.f, ss = 0.f;
  for (int i = blockIdx.x * blockDim.x + threadIdx.x; i < n; i += gridDim.x * blockDim.x) {
    float v = x[i]; s += v; ss += v * v;
  }
#pragma unroll
  for (int o = 16; o > 0; o >>= 1) { s += __shfl_down(s, o); ss += __shfl_down(ss, o); }
  if ((threadIdx.x & 31) == 0) { atomicAdd(&st[0], s); atomicAdd(&st[1], ss); }
}

__global__ void k_bnapply(const float* x, _Float16* out, const float* st,
                          const float* g, const float* be, int n, int relu) {
  int i = blockIdx.x * blockDim.x + threadIdx.x;
  if (i >= n) return;
  float m = st[0] / n, var = st[1] / n - m * m;
  float inv = rsqrtf(var + BN_EPS);
  float y = (x[i] - m) * inv * g[0] + be[0];
  if (relu) y = fmaxf(y, 0.f);
  out[i] = (_Float16)y;
}

// ---------------- output: relu(bn(z2)[b2,h] + bn(p2)[b1,h]) broadcast -> (256,256,256) ----------------
__global__ void k_out(const float* z2, const float* p2, const float* st,
                      const float* gz, const float* bz, const float* gp, const float* bp,
                      float* out) {
  int i = blockIdx.x * 256 + threadIdx.x;
  int h = i & 255, b2 = (i >> 8) & 255, b1 = i >> 16;
  const float nn = 65536.f;
  float mz = st[2] / nn, vz = st[3] / nn - mz * mz, iz = rsqrtf(vz + BN_EPS);
  float mp = st[6] / nn, vp = st[7] / nn - mp * mp, ip = rsqrtf(vp + BN_EPS);
  float z = (z2[b2 * 256 + h] - mz) * iz * gz[0] + bz[0];
  float p = (p2[b1 * 256 + h] - mp) * ip * gp[0] + bp[0];
  out[i] = fmaxf(z + p, 0.f);
}

// ---------------- workspace layout ----------------
#define OFF_RISR  ((size_t)0)
#define OFF_RISI  (OFF_RISR + 256*1024*4)
#define OFF_XS    (OFF_RISI + 256*1024*4)
#define OFF_U4VR  (OFF_XS   + 256*4096*4)
#define OFF_U4VI  (OFF_U4VR + (size_t)4096*1024*2)
#define OFF_U4NR  (OFF_U4VI + (size_t)4096*1024*2)
#define OFF_U4NI  (OFF_U4NR + (size_t)4096*1024*2)
#define OFF_U3TR  (OFF_U4NI + (size_t)4096*1024*2)
#define OFF_U3TI  (OFF_U3TR + 16*4096*2)
#define OFF_U2TR  (OFF_U3TI + 16*4096*2)
#define OFF_U2TI  (OFF_U2TR + 16*1024*2)
#define OFF_FEAT  (OFF_U2TI + 16*1024*2)
#define OFF_FEATH (OFF_FEAT + 256*512*4)
#define OFF_WM1H  (OFF_FEATH+ 256*512*2)
#define OFF_WM2H  (OFF_WM1H + 512*512*2)
#define OFF_WP1H  (OFF_WM2H + 256*512*2)
#define OFF_WP2H  (OFF_WP1H + 512*1024*2)
#define OFF_PHH   (OFF_WP2H + 256*512*2)
#define OFF_Z1    (OFF_PHH  + 256*1024*2)
#define OFF_Z1H   (OFF_Z1   + 256*512*4)
#define OFF_Z2    (OFF_Z1H  + 256*512*2)
#define OFF_P1    (OFF_Z2   + 256*256*4)
#define OFF_P1H   (OFF_P1   + 256*512*4)
#define OFF_P2    (OFF_P1H  + 256*512*2)
#define OFF_STAT  (OFF_P2   + 256*256*4)

extern "C" void kernel_launch(void* const* d_in, const int* in_sizes, int n_in,
                              void* d_out, int out_size, void* d_ws, size_t ws_size,
                              hipStream_t stream) {
  (void)in_sizes; (void)n_in; (void)out_size; (void)ws_size;
  const float*  x     = (const float*)d_in[0];
  const float*  phase = (const float*)d_in[1];
  const float2* uc0   = (const float2*)d_in[2];
  const float2* uc1   = (const float2*)d_in[3];
  const float2* uc2   = (const float2*)d_in[4];
  const float2* uc3   = (const float2*)d_in[5];
  const float2* uc4   = (const float2*)d_in[6];
  const float2* uc5   = (const float2*)d_in[7];
  const float *Wm1=(const float*)d_in[8],  *bm1=(const float*)d_in[9],
              *gm1=(const float*)d_in[10], *bem1=(const float*)d_in[11];
  const float *Wm2=(const float*)d_in[12], *bm2=(const float*)d_in[13],
              *gm2=(const float*)d_in[14], *bem2=(const float*)d_in[15];
  const float *Wp1=(const float*)d_in[16], *bp1=(const float*)d_in[17],
              *gp1=(const float*)d_in[18], *bep1=(const float*)d_in[19];
  const float *Wp2=(const float*)d_in[20], *bp2=(const float*)d_in[21],
              *gp2=(const float*)d_in[22], *bep2=(const float*)d_in[23];

  char* ws = (char*)d_ws;
  float    *ris_r=(float*)(ws+OFF_RISR), *ris_i=(float*)(ws+OFF_RISI), *xs=(float*)(ws+OFF_XS);
  _Float16 *u4vr=(_Float16*)(ws+OFF_U4VR), *u4vi=(_Float16*)(ws+OFF_U4VI);
  _Float16 *u4nr=(_Float16*)(ws+OFF_U4NR), *u4ni=(_Float16*)(ws+OFF_U4NI);
  _Float16 *u3tr=(_Float16*)(ws+OFF_U3TR), *u3ti=(_Float16*)(ws+OFF_U3TI);
  _Float16 *u2tr=(_Float16*)(ws+OFF_U2TR), *u2ti=(_Float16*)(ws+OFF_U2TI);
  float    *feat=(float*)(ws+OFF_FEAT);
  _Float16 *feath=(_Float16*)(ws+OFF_FEATH);
  _Float16 *wm1h=(_Float16*)(ws+OFF_WM1H), *wm2h=(_Float16*)(ws+OFF_WM2H);
  _Float16 *wp1h=(_Float16*)(ws+OFF_WP1H), *wp2h=(_Float16*)(ws+OFF_WP2H);
  _Float16 *phh=(_Float16*)(ws+OFF_PHH);
  float *z1=(float*)(ws+OFF_Z1), *z2=(float*)(ws+OFF_Z2), *p1=(float*)(ws+OFF_P1), *p2=(float*)(ws+OFF_P2);
  _Float16 *z1h=(_Float16*)(ws+OFF_Z1H), *p1h=(_Float16*)(ws+OFF_P1H);
  float *stats=(float*)(ws+OFF_STAT);   // [0,1]=z1 [2,3]=z2 [4,5]=p1 [6,7]=p2

  // packs + init
  k_init<<<256, 256, 0, stream>>>(uc5, feat, stats);
  k_ris<<<1024, 256, 0, stream>>>(phase, ris_r, ris_i, 256*1024);
  k_xs<<<4096, 256, 0, stream>>>(x, xs, 256*4096);
  k_pack_u4<<<16384, 256, 0, stream>>>(uc4, u4vr, u4vi, u4nr, u4ni);
  k_pack_bt<<<256, 256, 0, stream>>>(uc3, u3tr, u3ti, 4096);
  k_pack_bt<<<64, 256, 0, stream>>>(uc2, u2tr, u2ti, 1024);
  k_f32h<<<1024, 256, 0, stream>>>(Wm1, wm1h, 512*512);
  k_f32h<<<512, 256, 0, stream>>>(Wm2, wm2h, 256*512);
  k_f32h<<<2048, 256, 0, stream>>>(Wp1, wp1h, 512*1024);
  k_f32h<<<512, 256, 0, stream>>>(Wp2, wp2h, 256*512);
  k_f32h<<<1024, 256, 0, stream>>>(phase, phh, 256*1024);

  // big complex WMMA stages (all four channel terms -> feat)
  k_s1<<<dim3(16, 256), 256, 0, stream>>>(uc0, uc1, ris_r, ris_i, xs, u4vr, u4vi, u3tr, u3ti, feat);
  k_s2<<<dim3(4, 256), 256, 0, stream>>>(uc0, uc1, ris_r, ris_i, xs, u4nr, u4ni, u2tr, u2ti, feat);
  k_f32h<<<512, 256, 0, stream>>>(feat, feath, 256*512);

  // measurement MLP
  k_gemm<<<dim3(16, 4), 256, 0, stream>>>(feath, wm1h, bm1, z1, 512, 512);
  k_bnstats<<<256, 256, 0, stream>>>(z1, 256*512, stats + 0);
  k_bnapply<<<512, 256, 0, stream>>>(z1, z1h, stats + 0, gm1, bem1, 256*512, 1);
  k_gemm<<<dim3(16, 2), 256, 0, stream>>>(z1h, wm2h, bm2, z2, 256, 512);
  k_bnstats<<<256, 256, 0, stream>>>(z2, 256*256, stats + 2);

  // phase MLP
  k_gemm<<<dim3(16, 4), 256, 0, stream>>>(phh, wp1h, bp1, p1, 512, 1024);
  k_bnstats<<<256, 256, 0, stream>>>(p1, 256*512, stats + 4);
  k_bnapply<<<512, 256, 0, stream>>>(p1, p1h, stats + 4, gp1, bep1, 256*512, 1);
  k_gemm<<<dim3(16, 2), 256, 0, stream>>>(p1h, wp2h, bp2, p2, 256, 512);
  k_bnstats<<<256, 256, 0, stream>>>(p2, 256*256, stats + 6);

  // broadcast-add output (256,256,256)
  k_out<<<65536, 256, 0, stream>>>(z2, p2, stats, gm2, bem2, gp2, bep2, (float*)d_out);
}
